// Spatial_Adapter_78091095376426
// MI455X (gfx1250) — matrix operand order repaired and checked by reading, not compiled, not run on hardware
//
#include <hip/hip_runtime.h>

typedef __attribute__((ext_vector_type(16))) _Float16 v16h;
typedef __attribute__((ext_vector_type(8)))  _Float16 v8h;
typedef __attribute__((ext_vector_type(8)))  float    v8f;
typedef __attribute__((ext_vector_type(4)))  float    v4f;

// ---------------------------------------------------------------- WMMA core
static __device__ __forceinline__ v8f wmma32(v16h a, v16h b, v8f c) {
  // D = A(16x32 f16) * B(32x16 f16) + C(16x16 f32)
  return __builtin_amdgcn_wmma_f32_16x16x32_f16(false, a, false, b, (short)0, c,
                                                false, false);
}

// A fragment: 16x32 f16 tile, row-major, leading dim ld.
// Lane l: row m=l&15, half hi=l>>4. VGPR0-3 hold K=hi*8..hi*8+7,
// VGPR4-7 hold K=16+hi*8..16+hi*8+7  ->  two contiguous 16B loads.
static __device__ __forceinline__ v16h frag_a_h(const _Float16* A, int ld) {
  const int lane = threadIdx.x & 31;
  const int m = lane & 15, hi = lane >> 4;
  v8h lo = *(const v8h*)(A + (size_t)m * ld + hi * 8);
  v8h hh = *(const v8h*)(A + (size_t)m * ld + 16 + hi * 8);
  return __builtin_shufflevector(lo, hh, 0, 1, 2, 3, 4, 5, 6, 7, 8, 9, 10, 11,
                                 12, 13, 14, 15);
}

// Same A fragment but converting from an f32 source on the fly.
static __device__ __forceinline__ v16h frag_a_f(const float* A, int ld) {
  const int lane = threadIdx.x & 31;
  const int m = lane & 15, hi = lane >> 4;
  const float* p0 = A + (size_t)m * ld + hi * 8;
  v4f a0 = *(const v4f*)(p0);
  v4f a1 = *(const v4f*)(p0 + 4);
  v4f a2 = *(const v4f*)(p0 + 16);
  v4f a3 = *(const v4f*)(p0 + 20);
  v16h r;
#pragma unroll
  for (int i = 0; i < 4; ++i) {
    r[i]      = (_Float16)a0[i];
    r[4 + i]  = (_Float16)a1[i];
    r[8 + i]  = (_Float16)a2[i];
    r[12 + i] = (_Float16)a3[i];
  }
  return r;
}

// B fragment: 32x16, taken from row-major weight W[n][k] (y = x @ W^T).
// Lane l: col n=l&15, half hi=l>>4; elements 0-7 K = hi*8..hi*8+7,
// elements 8-15 K = 16+hi*8..16+hi*8+7 (same as A).
static __device__ __forceinline__ v16h frag_b_h(const _Float16* W, int ldk) {
  const int lane = threadIdx.x & 31;
  const int n = lane & 15, hi = lane >> 4;
  v8h lo = *(const v8h*)(W + (size_t)n * ldk + hi * 8);
  v8h hh = *(const v8h*)(W + (size_t)n * ldk + 16 + hi * 8);
  return __builtin_shufflevector(lo, hh, 0, 1, 2, 3, 4, 5, 6, 7, 8, 9, 10, 11,
                                 12, 13, 14, 15);
}

// 16 rows x 192 cols (12 tiles) panel accumulate, f16 A and W.
static __device__ __forceinline__ void gemm12_h(const _Float16* A, int lda,
                                                const _Float16* W, int K,
                                                v8f acc[12]) {
  for (int k = 0; k < K; k += 32) {
    v16h a = frag_a_h(A + k, lda);
#pragma unroll
    for (int t = 0; t < 12; ++t)
      acc[t] = wmma32(a, frag_b_h(W + (size_t)(t * 16) * K + k, K), acc[t]);
  }
}

// ------------------------------------------------ K0: f32 -> f16 weight cast
__global__ void k_cvt(const float* __restrict__ s, _Float16* __restrict__ d,
                      int n) {
  int i = blockIdx.x * 256 + threadIdx.x;
  if (i < n) d[i] = (_Float16)s[i];
}

// ---- K1: fused fc1 -> convA -> convB -> +residual -> GELU -> LayerNorm ----
__global__ __launch_bounds__(256) void k_front(
    const float* __restrict__ x, const _Float16* __restrict__ w1,
    const float* __restrict__ b1, const _Float16* __restrict__ wa,
    const float* __restrict__ ba, const _Float16* __restrict__ wb,
    const float* __restrict__ bb, const float* __restrict__ lng,
    const float* __restrict__ lnb, float* __restrict__ xs_out,
    _Float16* __restrict__ ln_out) {
  __shared__ _Float16 smem[8 * 16 * 192];  // 48 KB, wave-private regions
  const int wave = threadIdx.x >> 5;
  const int lane = threadIdx.x & 31;
  const int row0 = (blockIdx.x * 8 + wave) * 16;
  _Float16* stage = smem + wave * (16 * 192);

  // fc1: xs0 = x @ w1^T + b1   (K=768, N=192)
  v8f acc1[12] = {};
  for (int k = 0; k < 768; k += 32) {
    v16h a = frag_a_f(x + (size_t)row0 * 768 + k, 768);
#pragma unroll
    for (int t = 0; t < 12; ++t)
      acc1[t] = wmma32(a, frag_b_h(w1 + (size_t)(t * 16) * 768 + k, 768),
                       acc1[t]);
  }
  // bias; keep f32 in regs (residual) + stage f16 row-major copy in LDS.
#pragma unroll
  for (int t = 0; t < 12; ++t) {
    const int n = t * 16 + (lane & 15);
    const float bn = b1[n];
#pragma unroll
    for (int j = 0; j < 8; ++j) {
      const int m = j + (lane >> 4) * 8;
      float v = acc1[t][j] + bn;
      acc1[t][j] = v;
      stage[m * 192 + n] = (_Float16)v;
    }
  }
  asm volatile("s_wait_dscnt 0x0" ::: "memory");  // same-wave LDS RAW fence

  // convA: a = xs0 @ wa^T + ba   (K=192, N=64)
  v8f acc2[4] = {};
#pragma unroll
  for (int k = 0; k < 192; k += 32) {
    v16h a = frag_a_h(stage + k, 192);
#pragma unroll
    for (int t = 0; t < 4; ++t)
      acc2[t] = wmma32(a, frag_b_h(wa + (size_t)(t * 16) * 192 + k, 192),
                       acc2[t]);
  }
  // stage 'a' (16x64 f16) in front of region (xs0 copy fully consumed;
  // same-wave LDS ops are processed in order).
#pragma unroll
  for (int t = 0; t < 4; ++t) {
    const int n = t * 16 + (lane & 15);
    const float bn = ba[n];
#pragma unroll
    for (int j = 0; j < 8; ++j) {
      const int m = j + (lane >> 4) * 8;
      stage[m * 64 + n] = (_Float16)(acc2[t][j] + bn);
    }
  }
  asm volatile("s_wait_dscnt 0x0" ::: "memory");

  // convB: b_ = a @ wb^T + bb ; xs = gelu(xs0 + b_) ; ln = LN(xs)
  v8f acc3[12] = {};
#pragma unroll
  for (int k = 0; k < 64; k += 32) {
    v16h a = frag_a_h(stage + k, 64);
#pragma unroll
    for (int t = 0; t < 12; ++t)
      acc3[t] = wmma32(a, frag_b_h(wb + (size_t)(t * 16) * 64 + k, 64),
                       acc3[t]);
  }
#pragma unroll
  for (int t = 0; t < 12; ++t) {
    const int n = t * 16 + (lane & 15);
    const float bn = bb[n];
#pragma unroll
    for (int j = 0; j < 8; ++j) {
      const int m = j + (lane >> 4) * 8;
      float v = acc1[t][j] + acc3[t][j] + bn;
      float gl = 0.5f * v * (1.0f + erff(v * 0.70710678118654752f));
      acc3[t][j] = gl;
      xs_out[(size_t)(row0 + m) * 192 + n] = gl;
    }
  }
  // Fused LayerNorm: each half-wave holds full 192-ch rows -> reduce over the
  // 16 lanes of a half (xor masks 1,2,4,8 only; mask 16 would cross rows).
#pragma unroll
  for (int j = 0; j < 8; ++j) {
    float s = 0.f, ss = 0.f;
#pragma unroll
    for (int t = 0; t < 12; ++t) {
      float gv = acc3[t][j];
      s += gv;
      ss += gv * gv;
    }
#pragma unroll
    for (int msk = 1; msk <= 8; msk <<= 1) {
      s  += __shfl_xor(s, msk, 32);
      ss += __shfl_xor(ss, msk, 32);
    }
    const float mean = s * (1.0f / 192.0f);
    const float var  = ss * (1.0f / 192.0f) - mean * mean;
    const float rstd = rsqrtf(var + 1e-5f);
    const int m = j + (lane >> 4) * 8;
#pragma unroll
    for (int t = 0; t < 12; ++t) {
      const int n = t * 16 + (lane & 15);
      float lv = (acc3[t][j] - mean) * rstd * lng[n] + lnb[n];
      ln_out[(size_t)(row0 + m) * 192 + n] = (_Float16)lv;
    }
  }
}

// ------------------------- K2: qkv = ln @ qkv_w^T + b (q pre-scaled) -------
__global__ __launch_bounds__(256) void k_qkv(const _Float16* __restrict__ ln,
                                             const _Float16* __restrict__ w,
                                             const float* __restrict__ b,
                                             _Float16* __restrict__ qkv) {
  const int wave = threadIdx.x >> 5, lane = threadIdx.x & 31;
  const int row0 = (blockIdx.x * 8 + wave) * 16;
  const int col0 = blockIdx.y * 192;
  v8f acc[12] = {};
  gemm12_h(ln + (size_t)row0 * 192, 192, w + (size_t)col0 * 192, 192, acc);
  const float scale = (col0 == 0) ? 0.14433756729740643f : 1.0f;  // 48^-0.5
#pragma unroll
  for (int t = 0; t < 12; ++t) {
    const int n = col0 + t * 16 + (lane & 15);
    const float bn = b[n];
#pragma unroll
    for (int j = 0; j < 8; ++j) {
      const int m = j + (lane >> 4) * 8;
      qkv[(size_t)(row0 + m) * 576 + n] = (_Float16)((acc[t][j] + bn) * scale);
    }
  }
}

// ---------------------- K3: 5x5 neighborhood attention (hd=48, NH=4) -------
__global__ __launch_bounds__(256) void k_attn(const _Float16* __restrict__ qkv,
                                              const float* __restrict__ rpb,
                                              _Float16* __restrict__ out) {
  const int tid = blockIdx.x * 256 + threadIdx.x;
  const int head = tid & 3;
  const int p = tid >> 2;                 // token row index r = l*64 + bt
  const int bt = p & 63, l = p >> 6;
  const int iy = l >> 5, ix = l & 31;
  const int y0 = min(max(iy - 2, 0), 27);
  const int x0 = min(max(ix - 2, 0), 27);
  const _Float16* qp = qkv + (size_t)p * 576 + head * 48;
  v8h q[6];
#pragma unroll
  for (int c = 0; c < 6; ++c) q[c] = *(const v8h*)(qp + c * 8);
  float logits[25];
  float mx = -3.0e38f;
#pragma unroll
  for (int dy = 0; dy < 5; ++dy) {
#pragma unroll
    for (int dx = 0; dx < 5; ++dx) {
      const int ny = y0 + dy, nx = x0 + dx;
      const int rn = ((ny * 32 + nx) << 6) + bt;
      const _Float16* kp = qkv + (size_t)rn * 576 + 192 + head * 48;
      float s = 0.f;
#pragma unroll
      for (int c = 0; c < 6; ++c) {
        v8h kv = *(const v8h*)(kp + c * 8);
#pragma unroll
        for (int e = 0; e < 8; ++e) s += (float)q[c][e] * (float)kv[e];
      }
      s += rpb[(head * 9 + (ny - iy + 4)) * 9 + (nx - ix + 4)];
      logits[dy * 5 + dx] = s;
      mx = fmaxf(mx, s);
    }
  }
  float se = 0.f;
#pragma unroll
  for (int i = 0; i < 25; ++i) {
    logits[i] = __expf(logits[i] - mx);
    se += logits[i];
  }
  const float inv = 1.0f / se;
  float acc[48];
#pragma unroll
  for (int c = 0; c < 48; ++c) acc[c] = 0.f;
#pragma unroll
  for (int dy = 0; dy < 5; ++dy) {
#pragma unroll
    for (int dx = 0; dx < 5; ++dx) {
      const int ny = y0 + dy, nx = x0 + dx;
      const int rn = ((ny * 32 + nx) << 6) + bt;
      const _Float16* vp = qkv + (size_t)rn * 576 + 384 + head * 48;
      const float wgt = logits[dy * 5 + dx] * inv;
#pragma unroll
      for (int c = 0; c < 6; ++c) {
        v8h vv = *(const v8h*)(vp + c * 8);
#pragma unroll
        for (int e = 0; e < 8; ++e) acc[c * 8 + e] += wgt * (float)vv[e];
      }
    }
  }
  _Float16* op = out + (size_t)p * 192 + head * 48;
#pragma unroll
  for (int c = 0; c < 48; ++c) op[c] = (_Float16)acc[c];
}

// -------------------- K4: proj + bias + gelu-branch residual ---------------
__global__ __launch_bounds__(256) void k_proj(const _Float16* __restrict__ a,
                                              const _Float16* __restrict__ w,
                                              const float* __restrict__ b,
                                              const float* __restrict__ resid,
                                              _Float16* __restrict__ y) {
  const int wave = threadIdx.x >> 5, lane = threadIdx.x & 31;
  const int row0 = (blockIdx.x * 8 + wave) * 16;
  v8f acc[12] = {};
  gemm12_h(a + (size_t)row0 * 192, 192, w, 192, acc);
#pragma unroll
  for (int t = 0; t < 12; ++t) {
    const int n = t * 16 + (lane & 15);
    const float bn = b[n];
#pragma unroll
    for (int j = 0; j < 8; ++j) {
      const int m = j + (lane >> 4) * 8;
      const size_t idx = (size_t)(row0 + m) * 192 + n;
      y[idx] = (_Float16)(acc[t][j] + bn + resid[idx]);
    }
  }
}

// ---------- K5: fc2 + bias + input residual + final permutation scatter ----
// M-blocking here picks 16 rows with the SAME bt and 16 CONSECUTIVE l values
// (token stride 64), so a lane's 8 accumulator elements land on contiguous
// output addresses -> two global_store_b128 per tile instead of 8 scattered
// b32 stores on the dominant 201 MB output stream. The x-residual reads also
// become lane-consecutive (fully coalesced).
__global__ __launch_bounds__(256) void k_fc2(const _Float16* __restrict__ a,
                                             const _Float16* __restrict__ w,
                                             const float* __restrict__ b,
                                             const float* __restrict__ x,
                                             float* __restrict__ out) {
  const int wave = threadIdx.x >> 5, lane = threadIdx.x & 31;
  const int wb = blockIdx.x * 8 + wave;  // 4096 row-blocks
  const int bt = wb & 63;                // fixed batch-token for this block
  const int l0 = (wb >> 6) * 16;         // 16 consecutive spatial positions
  const int col0 = blockIdx.y * 192;
  const size_t r0 = (size_t)l0 * 64 + bt;  // rows r0, r0+64, ... r0+15*64
  v8f acc[12] = {};
  for (int k = 0; k < 192; k += 32) {
    v16h af = frag_a_h(a + r0 * 192 + k, 64 * 192);  // lda = 64 tokens
#pragma unroll
    for (int t = 0; t < 12; ++t)
      acc[t] = wmma32(
          af, frag_b_h(w + (size_t)(col0 + t * 16) * 192 + k, 192), acc[t]);
  }
  const int g = bt >> 4, tt = bt & 15;
  const int hi = lane >> 4;
#pragma unroll
  for (int t = 0; t < 12; ++t) {
    const int n = col0 + t * 16 + (lane & 15);  // d in [0,768)
    const float bn = b[n];
    // out[g][d][tt][l] : this lane's 8 elements are l0+hi*8 .. +7 (contiguous)
    float* op = out + ((size_t)(g * 768 + n) * 16 + tt) * 1024 + l0 + hi * 8;
    v4f v0, v1;
#pragma unroll
    for (int j = 0; j < 8; ++j) {
      const size_t r = r0 + (size_t)(j + hi * 8) * 64;
      const float val = acc[t][j] + bn + x[r * 768 + n];
      if (j < 4) v0[j] = val;
      else       v1[j - 4] = val;
    }
    *(v4f*)op       = v0;
    *(v4f*)(op + 4) = v1;
  }
}

// ---------------------------------------------------------------------------
extern "C" void kernel_launch(void* const* d_in, const int* in_sizes, int n_in,
                              void* d_out, int out_size, void* d_ws,
                              size_t ws_size, hipStream_t stream) {
  (void)in_sizes; (void)n_in; (void)out_size; (void)ws_size;
  const float* x       = (const float*)d_in[0];
  const float* fc1_w   = (const float*)d_in[1];
  const float* fc1_b   = (const float*)d_in[2];
  const float* convA_w = (const float*)d_in[3];
  const float* convA_b = (const float*)d_in[4];
  const float* convB_w = (const float*)d_in[5];
  const float* convB_b = (const float*)d_in[6];
  const float* ln_g    = (const float*)d_in[7];
  const float* ln_b    = (const float*)d_in[8];
  const float* qkv_w   = (const float*)d_in[9];
  const float* qkv_b   = (const float*)d_in[10];
  const float* rpb     = (const float*)d_in[11];
  const float* proj_w  = (const float*)d_in[12];
  const float* proj_b  = (const float*)d_in[13];
  const float* fc2_w   = (const float*)d_in[14];
  const float* fc2_b   = (const float*)d_in[15];

  char* ws = (char*)d_ws;
  size_t off = 0;
  auto alloc = [&](size_t bytes) {
    size_t o = off;
    off += (bytes + 255) & ~(size_t)255;
    return o;
  };
  const size_t M = 65536;  // L * BT tokens (row index r = l*64 + bt)
  _Float16* w1h  = (_Float16*)(ws + alloc(192 * 768 * 2));
  _Float16* wah  = (_Float16*)(ws + alloc(64 * 192 * 2));
  _Float16* wbh  = (_Float16*)(ws + alloc(192 * 64 * 2));
  _Float16* wqh  = (_Float16*)(ws + alloc(576 * 192 * 2));
  _Float16* wph  = (_Float16*)(ws + alloc(192 * 192 * 2));
  _Float16* w2h  = (_Float16*)(ws + alloc(768 * 192 * 2));
  float*    xsf  = (float*)   (ws + alloc(M * 192 * 4));  // gelu branch (f32)
  _Float16* lnh  = (_Float16*)(ws + alloc(M * 192 * 2));  // LN output
  _Float16* qkvh = (_Float16*)(ws + alloc(M * 576 * 2));  // q|k|v packed
  _Float16* ath  = (_Float16*)(ws + alloc(M * 192 * 2));  // attn out
  _Float16* yh   = (_Float16*)(ws + alloc(M * 192 * 2));  // proj+resid

  auto cvt = [&](const float* s, _Float16* d, int n) {
    k_cvt<<<(n + 255) / 256, 256, 0, stream>>>(s, d, n);
  };
  cvt(fc1_w,  w1h, 192 * 768);
  cvt(convA_w, wah, 64 * 192);
  cvt(convB_w, wbh, 192 * 64);
  cvt(qkv_w,  wqh, 576 * 192);
  cvt(proj_w, wph, 192 * 192);
  cvt(fc2_w,  w2h, 768 * 192);

  k_front<<<512, 256, 0, stream>>>(x, w1h, fc1_b, wah, convA_b, wbh, convB_b,
                                   ln_g, ln_b, xsf, lnh);
  k_qkv<<<dim3(512, 3), 256, 0, stream>>>(lnh, wqh, qkv_b, qkvh);
  k_attn<<<1024, 256, 0, stream>>>(qkvh, rpb, ath);
  k_proj<<<512, 256, 0, stream>>>(ath, wph, proj_b, xsf, yh);
  k_fc2<<<dim3(512, 4), 256, 0, stream>>>(yh, w2h, fc2_b, x, (float*)d_out);
}
